// Expert_Gate_69337952027069
// MI455X (gfx1250) — matrix operand order repaired and checked by reading, not compiled, not run on hardware
//
#include <hip/hip_runtime.h>
#include <hip/hip_bf16.h>

// ---------------- types ----------------
typedef __attribute__((ext_vector_type(16))) __bf16 bf16x16;
typedef __attribute__((ext_vector_type(8)))  float  f32x8;
typedef __attribute__((ext_vector_type(4)))  float  f32x4;
typedef __attribute__((ext_vector_type(4)))  int    i32x4;

union Frag {
  bf16x16 bf;
  i32x4   i4[2];
};

#define AS1 __attribute__((address_space(1)))
#define AS3 __attribute__((address_space(3)))

// Async global->LDS staging (CDNA5 ASYNCcnt path), compile-safe gated.
#if defined(__has_builtin)
#if __has_builtin(__builtin_amdgcn_global_load_async_to_lds_b128)
#define HAVE_ASYNC_LDS 1
#endif
#endif
#ifndef HAVE_ASYNC_LDS
#define HAVE_ASYNC_LDS 0
#endif

#if HAVE_ASYNC_LDS
#if __has_builtin(__builtin_amdgcn_s_wait_asynccnt)
#define WAIT_ASYNC(n) __builtin_amdgcn_s_wait_asynccnt(n)
#else
#define WAIT_ASYNC(n) asm volatile("s_wait_asynccnt %0" ::"i"(n) : "memory")
#endif
#endif

// Problem constants
#define BB 16384
#define II 512
#define H1 256
#define H2 128
#define EE 8
#define TT 2

// ws layout (bytes)
#define W1P_OFF 0u
#define W2P_OFF 2097152u      // E*I*H1*2 = 2MB
#define WGP_OFF 2621440u      // +E*H1*H2*2 = 512KB

// ---------------- pack: f32 weights -> bf16 in WMMA B-fragment lane order ----------------
// 32x16 tile, 512 bf16: idx = ((tile)*32 + lane)*16 + s ; element (K,N) with
// N = nb*16 + (lane&15), K = kb*32 + ((s>=8)?16:0) + ((lane>=16)?8:0) + (s&7)
__global__ __launch_bounds__(256) void pack_weights(
    const float* __restrict__ W1, const float* __restrict__ W2,
    const float* __restrict__ Wg,
    __bf16* __restrict__ W1p, __bf16* __restrict__ W2p,
    __bf16* __restrict__ Wgp)
{
  int i = blockIdx.x * blockDim.x + threadIdx.x;
  if (i < (1 << 20)) {              // W1p: e(3) kb(4) nb(4) l(5) s(4)
    int s = i & 15, l = (i >> 4) & 31, nb = (i >> 9) & 15, kb = (i >> 13) & 15, e = i >> 17;
    int k = kb * 32 + ((s >> 3) << 4) + ((l >> 4) << 3) + (s & 7);
    int n = nb * 16 + (l & 15);
    W1p[i] = (__bf16)W1[(e * II + k) * H1 + n];
  }
  if (i < (1 << 18)) {              // W2p: e(3) kb(3) nb(3) l(5) s(4)
    int s = i & 15, l = (i >> 4) & 31, nb = (i >> 9) & 7, kb = (i >> 12) & 7, e = i >> 15;
    int k = kb * 32 + ((s >> 3) << 4) + ((l >> 4) << 3) + (s & 7);
    int n = nb * 16 + (l & 15);
    W2p[i] = (__bf16)W2[(e * H1 + k) * H2 + n];
  }
  if (i < 8192) {                   // Wgp: kb(4) l(5) s(4); N = t*8+e
    int s = i & 15, l = (i >> 4) & 31, kb = i >> 9;
    int k = kb * 32 + ((s >> 3) << 4) + ((l >> 4) << 3) + (s & 7);
    int n = l & 15;
    Wgp[i] = (__bf16)Wg[((n >> 3) * II + k) * EE + (n & 7)];
  }
}

// ---------------- staging helpers (8KB chunk, 128 threads x 4 x 16B) ----------------
__device__ __forceinline__ void stage_fill(unsigned char* dstLds,
                                           const unsigned char* srcGlobal, int tid) {
#if HAVE_ASYNC_LDS
#pragma unroll
  for (int i = 0; i < 4; ++i) {
    i32x4* sgen = (i32x4*)(srcGlobal + (size_t)(tid + i * 128) * 16);  // strip const, retype
    i32x4* dgen = (i32x4*)(dstLds + (size_t)(tid + i * 128) * 16);
    __builtin_amdgcn_global_load_async_to_lds_b128(
        (AS1 i32x4*)sgen, (AS3 i32x4*)dgen, 0, 0);
  }
#else
  __builtin_prefetch((const char*)srcGlobal + 8192, 0, 1);
  const i32x4* s = (const i32x4*)srcGlobal;
  i32x4* d = (i32x4*)dstLds;
#pragma unroll
  for (int i = 0; i < 4; ++i) d[tid + i * 128] = s[tid + i * 128];
#endif
}
__device__ __forceinline__ void stage_drain_partial() {   // prev batch done, 1 batch in flight
#if HAVE_ASYNC_LDS
  WAIT_ASYNC(4);
#endif
}
__device__ __forceinline__ void stage_drain_all() {
#if HAVE_ASYNC_LDS
  WAIT_ASYNC(0);
#endif
}

// ---------------- fused MMoE kernel ----------------
// 4 waves/block, each wave owns a 16-row block of B.
// LDS: 4 * (8KB h + 1KB gates) + 2 * 8KB double-buffered weight stage = 52KB.
#define WAVES 4
#define PER_WAVE 9216
#define STAGE_OFF (PER_WAVE * WAVES)
#define SMEM_BYTES (STAGE_OFF + 2 * 8192)

__global__ __launch_bounds__(128) void moe_fused(
    const float* __restrict__ x,  const float* __restrict__ b1,
    const float* __restrict__ b2, const float* __restrict__ bg,
    const __bf16* __restrict__ W1p, const __bf16* __restrict__ W2p,
    const __bf16* __restrict__ Wgp, float* __restrict__ out)
{
  __shared__ __align__(16) unsigned char smem[SMEM_BYTES];
  const int tid  = threadIdx.x;
  const int lane = tid & 31;
  const int wave = tid >> 5;
  const int lrow = lane & 15;          // A-frag row / C-frag N / B-frag N
  const int lhi  = lane >> 4;          // lane half
  const int rb   = blockIdx.x * WAVES + wave;
  const int m0   = rb * 16;

  unsigned char* hbuf = smem + wave * PER_WAVE;   // 16 x 256 bf16 (8KB)
  float*         gbuf = (float*)(hbuf + 8192);    // gates [e][t][row] (1KB)
  float*         glog = (float*)hbuf;             // logits alias (used pre-expert-loop)
  unsigned char* stg[2] = { smem + STAGE_OFF, smem + STAGE_OFF + 8192 };

  f32x8 zero8;
#pragma unroll
  for (int r = 0; r < 8; ++r) zero8[r] = 0.0f;

  // ---- 1) load x rows once, convert (native v_cvt_pk_bf16_f32) to A-fragments in registers ----
  Frag xa[16];
  {
    const float* xrow = x + (long)(m0 + lrow) * II;
#pragma unroll
    for (int kb = 0; kb < 16; ++kb) {
      const float* p0 = xrow + kb * 32 + lhi * 8;  // K runs: {+0..7} and {+16..23}
      float v[16];
      *(f32x4*)(v + 0)  = *(const f32x4*)(p0);
      *(f32x4*)(v + 4)  = *(const f32x4*)(p0 + 4);
      *(f32x4*)(v + 8)  = *(const f32x4*)(p0 + 16);
      *(f32x4*)(v + 12) = *(const f32x4*)(p0 + 20);
#pragma unroll
      for (int s = 0; s < 16; ++s) xa[kb].bf[s] = (__bf16)v[s];
    }
  }

  // ---- 2) gate logits: one 16x16 WMMA chain over K=512 ----
  {
    f32x8 gacc = zero8;
#pragma unroll
    for (int kb = 0; kb < 16; ++kb) {
      Frag wb;
      const unsigned char* p = (const unsigned char*)Wgp + kb * 1024 + lane * 32;
      wb.i4[0] = *(const i32x4*)(p);
      wb.i4[1] = *(const i32x4*)(p + 16);
      gacc = __builtin_amdgcn_wmma_f32_16x16x32_bf16(false, xa[kb].bf, false, wb.bf,
                                                     (short)0, gacc, false, false);
    }
    float bgv = bg[lrow];                     // bg[T,E] flat == N index
#pragma unroll
    for (int r = 0; r < 8; ++r) {
      int m = r + lhi * 8;
      glog[m * 16 + lrow] = gacc[r] + bgv;    // logits [row][t*8+e]
    }
  }

  // ---- 3) softmax over experts; lane handles (row = lrow, t = lhi) ----
  {
    float l[8];
    *(f32x4*)(l)     = *(const f32x4*)(glog + lrow * 16 + lhi * 8);
    *(f32x4*)(l + 4) = *(const f32x4*)(glog + lrow * 16 + lhi * 8 + 4);
    float mx = l[0];
#pragma unroll
    for (int e = 1; e < 8; ++e) mx = fmaxf(mx, l[e]);
    float s = 0.0f;
#pragma unroll
    for (int e = 0; e < 8; ++e) { l[e] = __expf(l[e] - mx); s += l[e]; }
    float inv = 1.0f / s;
#pragma unroll
    for (int e = 0; e < 8; ++e) gbuf[(e * 2 + lhi) * 16 + lrow] = l[e] * inv;
  }

  // ---- 4) tower accumulators ----
  f32x8 tw0[8], tw1[8];
#pragma unroll
  for (int n = 0; n < 8; ++n) { tw0[n] = zero8; tw1[n] = zero8; }

  // ---- 5) expert loop ----
  for (int e = 0; e < EE; ++e) {
    // layer 1 in two H1 halves (caps accumulators at 64 VGPRs)
    for (int half = 0; half < 2; ++half) {
      const unsigned char* w1base = (const unsigned char*)W1p +
                                    (size_t)(((e * 16) * 16 + half * 8) * 512) * 2;
      // packed stride between k-steps: 16 tiles * 1KB
      f32x8 acc[8];
#pragma unroll
      for (int n = 0; n < 8; ++n) acc[n] = zero8;

      __syncthreads();                       // buffers free (prev chunk fully consumed)
      stage_fill(stg[0], w1base, tid);       // batch kb=0
#pragma unroll
      for (int kb = 0; kb < 16; ++kb) {
        if (kb + 1 < 16) {
          __syncthreads();                   // readers of buf[(kb+1)&1] (iter kb-1) done
          stage_fill(stg[(kb + 1) & 1], w1base + (size_t)(kb + 1) * 16384, tid);
          stage_drain_partial();             // batch kb resident, batch kb+1 in flight
        } else {
          stage_drain_all();
        }
        __syncthreads();                     // publish batch kb to all waves
        const unsigned char* sb = stg[kb & 1];
#pragma unroll
        for (int nbi = 0; nbi < 8; ++nbi) {
          Frag wb;
          const unsigned char* p = sb + nbi * 1024 + lane * 32;
          wb.i4[0] = *(const i32x4*)(p);
          wb.i4[1] = *(const i32x4*)(p + 16);
          acc[nbi] = __builtin_amdgcn_wmma_f32_16x16x32_bf16(false, xa[kb].bf, false, wb.bf,
                                                             (short)0, acc[nbi], false, false);
        }
      }
      // bias + relu -> h in wave-private LDS (row-major bf16 [16][256])
#pragma unroll
      for (int nbi = 0; nbi < 8; ++nbi) {
        int nb = half * 8 + nbi;
        float bv = b1[e * H1 + nb * 16 + lrow];
#pragma unroll
        for (int r = 0; r < 8; ++r) {
          float v = fmaxf(acc[nbi][r] + bv, 0.0f);
          int m = r + lhi * 8;
          *(__bf16*)(hbuf + m * 512 + (nb * 16 + lrow) * 2) = (__bf16)v;
        }
      }
    }

    // layer 2: e_out = relu(h @ W2[e] + b2[e])
    const unsigned char* w2base = (const unsigned char*)W2p + (size_t)(e * 8) * 8192;
    f32x8 acc2[8];
#pragma unroll
    for (int n = 0; n < 8; ++n) acc2[n] = zero8;

    __syncthreads();
    stage_fill(stg[0], w2base, tid);
#pragma unroll
    for (int kb2 = 0; kb2 < 8; ++kb2) {
      if (kb2 + 1 < 8) {
        __syncthreads();
        stage_fill(stg[(kb2 + 1) & 1], w2base + (size_t)(kb2 + 1) * 8192, tid);
        stage_drain_partial();
      } else {
        stage_drain_all();
      }
      __syncthreads();
      const unsigned char* sb = stg[kb2 & 1];
      Frag ha;   // A-fragment of h from LDS: two ds_load_b128
      const unsigned char* hp = hbuf + lrow * 512 + kb2 * 64 + lhi * 16;
      ha.i4[0] = *(const i32x4*)(hp);
      ha.i4[1] = *(const i32x4*)(hp + 32);
#pragma unroll
      for (int nb2 = 0; nb2 < 8; ++nb2) {
        Frag wb;
        const unsigned char* p = sb + nb2 * 1024 + lane * 32;
        wb.i4[0] = *(const i32x4*)(p);
        wb.i4[1] = *(const i32x4*)(p + 16);
        acc2[nb2] = __builtin_amdgcn_wmma_f32_16x16x32_bf16(false, ha.bf, false, wb.bf,
                                                            (short)0, acc2[nb2], false, false);
      }
    }

    // gate-weighted combine into towers; g[t][row] with row = r + lhi*8
    float g0[8], g1[8];
    *(f32x4*)(g0)     = *(const f32x4*)(gbuf + (e * 2 + 0) * 16 + lhi * 8);
    *(f32x4*)(g0 + 4) = *(const f32x4*)(gbuf + (e * 2 + 0) * 16 + lhi * 8 + 4);
    *(f32x4*)(g1)     = *(const f32x4*)(gbuf + (e * 2 + 1) * 16 + lhi * 8);
    *(f32x4*)(g1 + 4) = *(const f32x4*)(gbuf + (e * 2 + 1) * 16 + lhi * 8 + 4);
#pragma unroll
    for (int nb2 = 0; nb2 < 8; ++nb2) {
      float bv = b2[e * H2 + nb2 * 16 + lrow];
#pragma unroll
      for (int r = 0; r < 8; ++r) {
        float eo = fmaxf(acc2[nb2][r] + bv, 0.0f);
        tw0[nb2][r] += g0[r] * eo;
        tw1[nb2][r] += g1[r] * eo;
      }
    }
  }

  // ---- 6) write towers [T, B, H2] f32 ----
#pragma unroll
  for (int nb2 = 0; nb2 < 8; ++nb2) {
#pragma unroll
    for (int r = 0; r < 8; ++r) {
      int m = m0 + r + lhi * 8;
      int col = nb2 * 16 + lrow;
      out[(size_t)m * H2 + col]        = tw0[nb2][r];
      out[(size_t)(BB + m) * H2 + col] = tw1[nb2][r];
    }
  }
}

// ---------------- host launch ----------------
extern "C" void kernel_launch(void* const* d_in, const int* in_sizes, int n_in,
                              void* d_out, int out_size, void* d_ws, size_t ws_size,
                              hipStream_t stream) {
  (void)in_sizes; (void)n_in; (void)out_size; (void)ws_size;
  const float* x  = (const float*)d_in[0];
  const float* W1 = (const float*)d_in[1];
  const float* b1 = (const float*)d_in[2];
  const float* W2 = (const float*)d_in[3];
  const float* b2 = (const float*)d_in[4];
  const float* Wg = (const float*)d_in[5];
  const float* bg = (const float*)d_in[6];
  float* out = (float*)d_out;

  __bf16* W1p = (__bf16*)((char*)d_ws + W1P_OFF);
  __bf16* W2p = (__bf16*)((char*)d_ws + W2P_OFF);
  __bf16* Wgp = (__bf16*)((char*)d_ws + WGP_OFF);

  // repack weights to bf16 WMMA B-fragment layout (idempotent, reads only inputs)
  pack_weights<<<4096, 256, 0, stream>>>(W1, W2, Wg, W1p, W2p, Wgp);

  // fused MMoE: 1024 row-blocks of 16, 4 waves/block
  moe_fused<<<BB / (16 * WAVES), 32 * WAVES, 0, stream>>>(x, b1, b2, bg, W1p, W2p, Wgp, out);
}
